// GAttentionBlock_21131239096926
// MI455X (gfx1250) — matrix operand order repaired and checked by reading, not compiled, not run on hardware
//
#include <hip/hip_runtime.h>
#include <hip/hip_bf16.h>

typedef __bf16 bf16_t;
typedef __attribute__((ext_vector_type(16))) __bf16 v16bf;
typedef __attribute__((ext_vector_type(8)))  __bf16 v8bf;
typedef __attribute__((ext_vector_type(8)))  float  v8f;

constexpr int B_ = 8, C_ = 640, T_ = 1024, NH_ = 8, D_ = 80, DP_ = 96, G_ = 32, CPG_ = 20;

union V16 { v16bf v; v8bf h[2]; };

__device__ inline v8f wmma_bf16(v16bf a, v16bf b, v8f c) {
  // D = A(16x32 bf16) * B(32x16 bf16) + C(16x16 f32)
  return __builtin_amdgcn_wmma_f32_16x16x32_bf16(false, a, false, b, (short)0, c, false, false);
}

// ---------------- utility kernels ----------------
__global__ void cvt_bf16_kernel(const float* __restrict__ src, bf16_t* __restrict__ dst, int n) {
  int i = blockIdx.x * blockDim.x + threadIdx.x;
  if (i < n) dst[i] = (bf16_t)src[i];
}

__global__ void zero_u32_kernel(unsigned int* __restrict__ p, int nwords) {
  int i = blockIdx.x * blockDim.x + threadIdx.x;
  if (i < nwords) p[i] = 0u;
}

// ---------------- GroupNorm -> xnT[b][t][c] bf16 ----------------
__global__ void __launch_bounds__(256) groupnorm_kernel(const float* __restrict__ x,
                                                        const float* __restrict__ gw,
                                                        const float* __restrict__ gb,
                                                        bf16_t* __restrict__ xnT) {
  int bg = blockIdx.x;
  int b = bg / G_, g = bg % G_;
  const float* xp = x + (size_t)(b * C_ + g * CPG_) * T_;
  const int N = CPG_ * T_;

  float s = 0.f, ss = 0.f;
  for (int i = threadIdx.x; i < N; i += 256) {
    float v = xp[i];
    s += v; ss += v * v;
  }
  for (int m = 16; m >= 1; m >>= 1) { s += __shfl_xor(s, m); ss += __shfl_xor(ss, m); }
  __shared__ float red[16];
  int wv = threadIdx.x >> 5, ln = threadIdx.x & 31;
  if (ln == 0) { red[wv] = s; red[8 + wv] = ss; }
  __syncthreads();
  float S = 0.f, SS = 0.f;
  for (int i = 0; i < 8; i++) { S += red[i]; SS += red[8 + i]; }
  float mean = S / (float)N;
  float var  = SS / (float)N - mean * mean;
  float rstd = rsqrtf(var + 1e-5f);

  for (int i = threadIdx.x; i < N; i += 256) {
    int ci = i / T_, tt = i - ci * T_;
    int c = g * CPG_ + ci;
    float v = (xp[i] - mean) * rstd * gw[c] + gb[c];
    xnT[(size_t)(b * T_ + tt) * C_ + c] = (bf16_t)v;
  }
}

// ---------------- QKV GEMM: qkv[o,t] = qkv_w[o,:]·xn[:,t] + qkv_b ----------------
// grid (4, 60, 8); wave = 16x16 tile x 4 N-tiles; ping-pong K pipeline
__global__ void __launch_bounds__(256) qkv_kernel(const bf16_t* __restrict__ wq,
                                                  const float* __restrict__ qkvb,
                                                  const bf16_t* __restrict__ xnT,
                                                  bf16_t* __restrict__ qm,
                                                  bf16_t* __restrict__ km,
                                                  bf16_t* __restrict__ vTm) {
  int b = blockIdx.z;
  int w = threadIdx.x >> 5, lane = threadIdx.x & 31;
  int lm = lane & 15, kh = lane >> 4;
  int row0  = blockIdx.y * 32 + (w >> 2) * 16;
  int nbase = blockIdx.x * 256 + (w & 3) * 16;

  v8f acc[4];
  for (int u = 0; u < 4; u++) for (int e = 0; e < 8; e++) acc[u][e] = 0.f;

  const bf16_t* arow = wq + (size_t)(row0 + lm) * C_;
  const bf16_t* brow[4];
  for (int u = 0; u < 4; u++)
    brow[u] = xnT + (size_t)(b * T_ + nbase + u * 64 + lm) * C_ + kh * 16;

  constexpr int KC = C_ / 32;   // 20 (even)
  auto loadA = [&](V16& a, int kc) {
    a.h[0] = *(const v8bf*)(arow + kc * 32 + kh * 16);
    a.h[1] = *(const v8bf*)(arow + kc * 32 + kh * 16 + 8);
  };
  auto loadB = [&](v16bf* bb, int kc) {
    for (int u = 0; u < 4; u++) bb[u] = *(const v16bf*)(brow[u] + kc * 32);
  };

  V16 a0, a1;
  v16bf b0[4], b1[4];
  loadA(a0, 0); loadB(b0, 0);
#pragma unroll
  for (int kc = 0; kc < KC; kc += 2) {
    loadA(a1, kc + 1); loadB(b1, kc + 1);
    for (int u = 0; u < 4; u++) acc[u] = wmma_bf16(a0.v, b0[u], acc[u]);
    if (kc + 2 < KC) { loadA(a0, kc + 2); loadB(b0, kc + 2); }
    for (int u = 0; u < 4; u++) acc[u] = wmma_bf16(a1.v, b1[u], acc[u]);
  }

  float scale = rsqrtf(sqrtf((float)D_));   // 1/d^0.25, applied to q and k
  for (int u = 0; u < 4; u++) {
    int t = nbase + u * 64 + lm;
    for (int j = 0; j < 8; j++) {
      int o = row0 + j + kh * 8;
      float val = acc[u][j] + qkvb[o];
      int which = o / C_;
      int rem = o - which * C_;
      int h = rem / D_, dd = rem - h * D_;
      size_t bh = (size_t)(b * NH_ + h);
      if (which == 0)      qm[(bh * T_ + t) * DP_ + dd]  = (bf16_t)(val * scale);
      else if (which == 1) km[(bh * T_ + t) * DP_ + dd]  = (bf16_t)(val * scale);
      else                 vTm[(bh * DP_ + dd) * T_ + t] = (bf16_t)val;
    }
  }
}

// ---------------- fused attention: block per (b,h,32-row strip) ----------------
// LDS: 32x1024 bf16 score/prob strip (64 KB)
__global__ void __launch_bounds__(256) attn_kernel(const bf16_t* __restrict__ qm,
                                                   const bf16_t* __restrict__ km,
                                                   const bf16_t* __restrict__ vTm,
                                                   bf16_t* __restrict__ aT) {
  extern __shared__ bf16_t smem[];           // 32 x 1024 bf16
  __shared__ float rowscale[32];
  int rb = blockIdx.x;                       // 0..31 (32 rows each)
  int h = blockIdx.y, b = blockIdx.z;
  size_t bh = (size_t)(b * NH_ + h);
  const bf16_t* qb = qm  + bh * T_ * DP_;
  const bf16_t* kb = km  + bh * T_ * DP_;
  const bf16_t* vb = vTm + bh * DP_ * T_;
  int w = threadIdx.x >> 5, lane = threadIdx.x & 31;
  int lm = lane & 15, kh = lane >> 4;

  // phase 1: S = Q K^T (K-dim = 96, zero-padded)
  // each wave owns one nt stream (nt0 = w, step 8) and computes BOTH 16-row
  // tiles against the same B registers -> K panel read once per block.
  {
    int tr0 = rb * 32 + lm;                  // rows 0..15 of strip
    int tr1 = rb * 32 + 16 + lm;             // rows 16..31 of strip
    V16 a0[3], a1[3];
    for (int kc = 0; kc < 3; kc++) {
      a0[kc].h[0] = *(const v8bf*)(qb + (size_t)tr0 * DP_ + kc * 32 + kh * 16);
      a0[kc].h[1] = *(const v8bf*)(qb + (size_t)tr0 * DP_ + kc * 32 + kh * 16 + 8);
      a1[kc].h[0] = *(const v8bf*)(qb + (size_t)tr1 * DP_ + kc * 32 + kh * 16);
      a1[kc].h[1] = *(const v8bf*)(qb + (size_t)tr1 * DP_ + kc * 32 + kh * 16 + 8);
    }
    const bf16_t* bbase = kb + (size_t)lm * DP_ + kh * 16;
    auto loadKB = [&](v16bf* bb, int nt) {
      for (int kc = 0; kc < 3; kc++)
        bb[kc] = *(const v16bf*)(bbase + (size_t)(nt * 16) * DP_ + kc * 32);
    };
    auto doTile2 = [&](const v16bf* bb, int nt) {
      v8f c0, c1;
      for (int e = 0; e < 8; e++) { c0[e] = 0.f; c1[e] = 0.f; }
      for (int kc = 0; kc < 3; kc++) {
        c0 = wmma_bf16(a0[kc].v, bb[kc], c0);
        c1 = wmma_bf16(a1[kc].v, bb[kc], c1);
      }
      for (int j = 0; j < 8; j++) {
        smem[(j + kh * 8) * 1024 + nt * 16 + lm]      = (bf16_t)c0[j];
        smem[(16 + j + kh * 8) * 1024 + nt * 16 + lm] = (bf16_t)c1[j];
      }
    };
    v16bf p0[3], p1[3];
    loadKB(p0, w);
#pragma unroll
    for (int nt = w; nt < 64; nt += 16) {
      loadKB(p1, nt + 8);
      doTile2(p0, nt);
      if (nt + 16 < 64) loadKB(p0, nt + 16);
      doTile2(p1, nt + 8);
    }
  }
  __syncthreads();

  // phase 2: softmax (fp32 math, bf16 storage), 8 threads per row
  {
    int row = threadIdx.x >> 3, oct = threadIdx.x & 7;
    bf16_t* rp = smem + row * 1024 + oct * 128;
    float m = -3.4e38f;
    for (int i = 0; i < 128; i++) m = fmaxf(m, (float)rp[i]);
    for (int msk = 1; msk < 8; msk <<= 1) m = fmaxf(m, __shfl_xor(m, msk));
    float s = 0.f;
    for (int i = 0; i < 128; i++) {
      float e = __expf((float)rp[i] - m);
      rp[i] = (bf16_t)e;
      s += e;
    }
    for (int msk = 1; msk < 8; msk <<= 1) s += __shfl_xor(s, msk);
    if (oct == 0) rowscale[row] = 1.f / s;
  }
  __syncthreads();

  // phase 3: A = P V (32 x 80 real columns); wave w (<5) owns V columns
  // pnt = w and computes both 16-row tiles sharing one V stream.
  if (w < 5) {
    int dcol = w * 16 + lm;                  // < 80 always
    const bf16_t* vcol = vb + (size_t)dcol * T_ + kh * 16;
    const bf16_t* prow0 = smem + lm * 1024;
    const bf16_t* prow1 = smem + (16 + lm) * 1024;
    v8f acc0, acc1;
    for (int e = 0; e < 8; e++) { acc0[e] = 0.f; acc1[e] = 0.f; }
    v16bf vb0 = *(const v16bf*)(vcol);
    v16bf vb1;
#pragma unroll
    for (int kc = 0; kc < 32; kc += 2) {
      vb1 = *(const v16bf*)(vcol + (kc + 1) * 32);
      V16 pa;
      pa.h[0] = *(const v8bf*)(prow0 + kc * 32 + kh * 16);
      pa.h[1] = *(const v8bf*)(prow0 + kc * 32 + kh * 16 + 8);
      acc0 = wmma_bf16(pa.v, vb0, acc0);
      pa.h[0] = *(const v8bf*)(prow1 + kc * 32 + kh * 16);
      pa.h[1] = *(const v8bf*)(prow1 + kc * 32 + kh * 16 + 8);
      acc1 = wmma_bf16(pa.v, vb0, acc1);
      if (kc + 2 < 32) vb0 = *(const v16bf*)(vcol + (kc + 2) * 32);
      pa.h[0] = *(const v8bf*)(prow0 + (kc + 1) * 32 + kh * 16);
      pa.h[1] = *(const v8bf*)(prow0 + (kc + 1) * 32 + kh * 16 + 8);
      acc0 = wmma_bf16(pa.v, vb1, acc0);
      pa.h[0] = *(const v8bf*)(prow1 + (kc + 1) * 32 + kh * 16);
      pa.h[1] = *(const v8bf*)(prow1 + (kc + 1) * 32 + kh * 16 + 8);
      acc1 = wmma_bf16(pa.v, vb1, acc1);
    }
    for (int j = 0; j < 8; j++) {
      int r0 = j + kh * 8;
      int r1 = 16 + j + kh * 8;
      int t0 = rb * 32 + r0;
      int t1 = rb * 32 + r1;
      aT[(size_t)(b * T_ + t0) * C_ + h * D_ + dcol] = (bf16_t)(acc0[j] * rowscale[r0]);
      aT[(size_t)(b * T_ + t1) * C_ + h * D_ + dcol] = (bf16_t)(acc1[j] * rowscale[r1]);
    }
  }
}

// ---------------- proj GEMM + bias + residual ----------------
// grid (4, 20, 8); ping-pong K pipeline
__global__ void __launch_bounds__(256) proj_kernel(const bf16_t* __restrict__ pw,
                                                   const float* __restrict__ pb,
                                                   const bf16_t* __restrict__ aT,
                                                   const float* __restrict__ x,
                                                   float* __restrict__ out) {
  int b = blockIdx.z;
  int w = threadIdx.x >> 5, lane = threadIdx.x & 31;
  int lm = lane & 15, kh = lane >> 4;
  int row0  = blockIdx.y * 32 + (w >> 2) * 16;
  int nbase = blockIdx.x * 256 + (w & 3) * 16;

  v8f acc[4];
  for (int u = 0; u < 4; u++) for (int e = 0; e < 8; e++) acc[u][e] = 0.f;

  const bf16_t* arow = pw + (size_t)(row0 + lm) * C_;
  const bf16_t* brow[4];
  for (int u = 0; u < 4; u++)
    brow[u] = aT + (size_t)(b * T_ + nbase + u * 64 + lm) * C_ + kh * 16;

  constexpr int KC = C_ / 32;   // 20 (even)
  auto loadA = [&](V16& a, int kc) {
    a.h[0] = *(const v8bf*)(arow + kc * 32 + kh * 16);
    a.h[1] = *(const v8bf*)(arow + kc * 32 + kh * 16 + 8);
  };
  auto loadB = [&](v16bf* bb, int kc) {
    for (int u = 0; u < 4; u++) bb[u] = *(const v16bf*)(brow[u] + kc * 32);
  };

  V16 a0, a1;
  v16bf b0[4], b1[4];
  loadA(a0, 0); loadB(b0, 0);
#pragma unroll
  for (int kc = 0; kc < KC; kc += 2) {
    loadA(a1, kc + 1); loadB(b1, kc + 1);
    for (int u = 0; u < 4; u++) acc[u] = wmma_bf16(a0.v, b0[u], acc[u]);
    if (kc + 2 < KC) { loadA(a0, kc + 2); loadB(b0, kc + 2); }
    for (int u = 0; u < 4; u++) acc[u] = wmma_bf16(a1.v, b1[u], acc[u]);
  }

  for (int u = 0; u < 4; u++) {
    int t = nbase + u * 64 + lm;
    for (int j = 0; j < 8; j++) {
      int o = row0 + j + kh * 8;
      size_t idx = (size_t)(b * C_ + o) * T_ + t;
      out[idx] = x[idx] + pb[o] + acc[u][j];
    }
  }
}

extern "C" void kernel_launch(void* const* d_in, const int* in_sizes, int n_in,
                              void* d_out, int out_size, void* d_ws, size_t ws_size,
                              hipStream_t stream) {
  (void)in_sizes; (void)n_in; (void)out_size; (void)ws_size;
  const float* x      = (const float*)d_in[0];
  const float* norm_w = (const float*)d_in[1];
  const float* norm_b = (const float*)d_in[2];
  const float* qkv_w  = (const float*)d_in[3];
  const float* qkv_b  = (const float*)d_in[4];
  const float* proj_w = (const float*)d_in[5];
  const float* proj_b = (const float*)d_in[6];
  float* out = (float*)d_out;

  char* ws = (char*)d_ws;
  size_t off = 0;
  auto take = [&](size_t bytes) -> char* {
    char* p = ws + off;
    off += (bytes + 255) & ~(size_t)255;
    return p;
  };
  bf16_t* xnT   = (bf16_t*)take((size_t)B_ * T_ * C_ * 2);
  bf16_t* wqkv  = (bf16_t*)take((size_t)3 * C_ * C_ * 2);
  bf16_t* wproj = (bf16_t*)take((size_t)C_ * C_ * 2);
  bf16_t* qm    = (bf16_t*)take((size_t)B_ * NH_ * T_ * DP_ * 2);
  bf16_t* km    = (bf16_t*)take((size_t)B_ * NH_ * T_ * DP_ * 2);
  bf16_t* vTm   = (bf16_t*)take((size_t)B_ * NH_ * DP_ * T_ * 2);
  bf16_t* aT    = (bf16_t*)take((size_t)B_ * T_ * C_ * 2);

  { int n = 3 * C_ * C_; cvt_bf16_kernel<<<(n + 255) / 256, 256, 0, stream>>>(qkv_w, wqkv, n); }
  { int n = C_ * C_;     cvt_bf16_kernel<<<(n + 255) / 256, 256, 0, stream>>>(proj_w, wproj, n); }
  { // zero q/k/vT (contiguous) so the d-padding (80..95) contributes zeros
    int nw = (int)((size_t)3 * B_ * NH_ * T_ * DP_ * 2 / 4);
    zero_u32_kernel<<<(nw + 255) / 256, 256, 0, stream>>>((unsigned int*)qm, nw);
  }
  groupnorm_kernel<<<dim3(B_ * G_), 256, 0, stream>>>(x, norm_w, norm_b, xnT);
  qkv_kernel<<<dim3(4, 60, 8), 256, 0, stream>>>(wqkv, qkv_b, xnT, qm, km, vTm);
  attn_kernel<<<dim3(32, NH_, B_), 256, 32 * 1024 * 2, stream>>>(qm, km, vTm, aT);
  proj_kernel<<<dim3(4, 20, 8), 256, 0, stream>>>(wproj, proj_b, aT, x, out);
}
